// ShiftedCrossLocalAttention_7851200217264
// MI455X (gfx1250) — compile-verified, matrix operand order checked by reading, not run on hardware
//
#include <hip/hip_runtime.h>

// ---------------------------------------------------------------------------
// Shifted-window cross attention for MI455X (gfx1250, wave32, WMMA).
//   B=4, C=128, H=W=256, window=8 (N=64 tokens), shift=4, heads=4, hd=32.
// One window per 256-thread block (8 wave32s). Entire per-window pipeline in
// LDS (320 KB/WGP). f16 WMMA (v_wmma_f32_16x16x32_f16) with f32 accumulation.
// Weights staged global->LDS with CDNA5 async-LDS loads (overlapped with the
// shifted-window gather); output scattered with async LDS->global stores.
// ---------------------------------------------------------------------------

typedef __attribute__((ext_vector_type(16))) _Float16 v16h;
typedef __attribute__((ext_vector_type(8)))  _Float16 v8h;
typedef __attribute__((ext_vector_type(8)))  float    v8f;

#define WSZ    8
#define SHIFT_ 4
#define HEADS_ 4
#define C_     128
#define H_     256
#define W_     256
#define B_     4
#define N_     64          // tokens per window
#define HD_    32
#define NWIN_  1024        // (256/8)^2 windows per image

constexpr int SWD = 136;   // f16 stride, 128-wide rows (bank-padded)
constexpr int SVT = 72;    // f16 stride, V^T rows (64 tokens; 144B = 16B-aligned)
constexpr int SAT = 66;    // f16 stride, 64-wide attn-prob rows
constexpr int SSC = 65;    // f32 stride, 64-wide score rows (odd => bank spread)

constexpr size_t SZ_WMAT = (size_t)C_ * SWD * sizeof(_Float16);            // 34816
constexpr size_t SZ_TOK  = (size_t)N_ * SWD * sizeof(_Float16);            // 17408
constexpr size_t SZ_VT   = (size_t)C_ * SVT * sizeof(_Float16);            // 18432
constexpr size_t SZ_S    = (size_t)(HEADS_ * N_) * SSC * sizeof(float);    // 66560
constexpr size_t SZ_P    = (size_t)(HEADS_ * N_) * SAT * sizeof(_Float16); // 33792
constexpr size_t OFF_WQ  = 0;
constexpr size_t OFF_WK  = OFF_WQ + SZ_WMAT;
constexpr size_t OFF_WV  = OFF_WK + SZ_WMAT;
constexpr size_t OFF_WO  = OFF_WV + SZ_WMAT;
constexpr size_t OFF_X   = OFF_WO + SZ_WMAT;   // reused later as attn output O
constexpr size_t OFF_Y   = OFF_X + SZ_TOK;
constexpr size_t OFF_Q   = OFF_Y + SZ_TOK;
constexpr size_t OFF_K   = OFF_Q + SZ_TOK;
constexpr size_t OFF_VT  = OFF_K + SZ_TOK;     // V stored transposed [C][N]
constexpr size_t OFF_S   = OFF_VT + SZ_VT;     // scores f32; reused as out f32
constexpr size_t OFF_P   = OFF_S + SZ_S;       // softmax probs f16
constexpr size_t SMEM_TOTAL = OFF_P + SZ_P;    // 327680 B == 320 KiB exactly

// ---------------------------------------------------------------------------
// CDNA5 async global<->LDS DMA (ISA 10.x / 15.18.3, tracked by ASYNCcnt).
// LDS addresses are byte offsets within the workgroup LDS allocation
// (dynamic LDS starts at 0: this kernel declares no static __shared__).
// ---------------------------------------------------------------------------
__device__ inline void async_g2l_b128(unsigned lds_off, const void* gaddr) {
  asm volatile("global_load_async_to_lds_b128 %0, %1, off"
               :: "v"(lds_off), "v"((unsigned long long)(size_t)gaddr)
               : "memory");
}
__device__ inline void async_l2g_b32(const void* gaddr, unsigned lds_off) {
  asm volatile("global_store_async_from_lds_b32 %0, %1, off"
               :: "v"((unsigned long long)(size_t)gaddr), "v"(lds_off)
               : "memory");
}
__device__ inline void wait_asynccnt0() {
  asm volatile("s_wait_asynccnt 0x0" ::: "memory");
}

// ---------------------------------------------------------------------------
// Fragment loaders/stores per CDNA5 ISA 7.12.2 (wave32).
// A 16x32 f16: lane 0-15 -> M=lane, VGPR0-3 K={0..7}, VGPR4-7 K={16..23};
//              lane 16-31 -> same M, K={8..15} / K={24..31}.
// B 32x16 f16: same pattern, lane indexes N (B rows striped across lanes).
// C/D 16x16 f32: VGPR r -> M=r (lanes 0-15) / M=r+8 (lanes 16-31), N=lane%16.
// ---------------------------------------------------------------------------
__device__ inline v16h ldfrag_rm(const _Float16* base, int row0, int k0, int stride) {
  const int lane = threadIdx.x & 31;
  const int lr = lane & 15, hi = lane >> 4;
  const _Float16* p = base + (size_t)(row0 + lr) * stride + k0;
  v16h f;
#pragma unroll
  for (int v = 0; v < 4; ++v) {
    int ka = hi * 8 + 2 * v;        // K 0..15 halves
    int kb = 16 + hi * 8 + 2 * v;   // K 16..31 halves
    f[2 * v]         = p[ka];
    f[2 * v + 1]     = p[ka + 1];
    f[8 + 2 * v]     = p[kb];
    f[8 + 2 * v + 1] = p[kb + 1];
  }
  return f;
}

__device__ inline void stfrag_f16_bias(_Float16* base, int row0, int col0, int stride,
                                       v8f d, float scale, float bv) {
  const int lane = threadIdx.x & 31;
  const int lr = lane & 15, hi = lane >> 4;
  const int n = col0 + lr;
#pragma unroll
  for (int r = 0; r < 8; ++r)
    base[(size_t)(row0 + hi * 8 + r) * stride + n] = (_Float16)(d[r] * scale + bv);
}

// Transposed f16 store: D tile rows are tokens, cols are channels; store at
// [channel][token]. Each lane writes 8 contiguous halves (one aligned 16B op).
__device__ inline void stfrag_f16_T(_Float16* base, int row0, int col0, int strideT,
                                    v8f d, float bv) {
  const int lane = threadIdx.x & 31;
  const int lr = lane & 15, hi = lane >> 4;
  const int n = col0 + lr;                       // channel
  _Float16* p = base + (size_t)n * strideT + row0 + hi * 8;
  v8h o;
#pragma unroll
  for (int r = 0; r < 8; ++r) o[r] = (_Float16)(d[r] + bv);
  *(v8h*)p = o;                                  // 16B-aligned (SVT*2 % 16 == 0)
}

__device__ inline void stfrag_f16_nb(_Float16* base, int row0, int col0, int stride, v8f d) {
  const int lane = threadIdx.x & 31;
  const int lr = lane & 15, hi = lane >> 4;
  const int n = col0 + lr;
#pragma unroll
  for (int r = 0; r < 8; ++r)
    base[(size_t)(row0 + hi * 8 + r) * stride + n] = (_Float16)d[r];
}

__device__ inline void stfrag_f32_nb(float* base, int row0, int col0, int stride, v8f d) {
  const int lane = threadIdx.x & 31;
  const int lr = lane & 15, hi = lane >> 4;
  const int n = col0 + lr;
#pragma unroll
  for (int r = 0; r < 8; ++r)
    base[(size_t)(row0 + hi * 8 + r) * stride + n] = d[r];
}

__device__ inline void stfrag_f32_bias(float* base, int row0, int col0, int stride,
                                       v8f d, float bv) {
  const int lane = threadIdx.x & 31;
  const int lr = lane & 15, hi = lane >> 4;
  const int n = col0 + lr;
#pragma unroll
  for (int r = 0; r < 8; ++r)
    base[(size_t)(row0 + hi * 8 + r) * stride + n] = d[r] + bv;
}

__device__ inline v8f wmma_f16(v16h a, v16h b, v8f c) {
  return __builtin_amdgcn_wmma_f32_16x16x32_f16(false, a, false, b, (short)0, c,
                                                false, false);
}

// ---------------------------------------------------------------------------
// Pre-kernel: convert the four 128x128 f32 weight matrices to f16 in d_ws.
// ---------------------------------------------------------------------------
__global__ __launch_bounds__(256) void cvt_weights_kernel(
    const float* __restrict__ Wq, const float* __restrict__ Wk,
    const float* __restrict__ Wv, const float* __restrict__ Wo,
    _Float16* __restrict__ dst) {
  int i = blockIdx.x * 256 + threadIdx.x;
  if (i < C_ * C_) {
    dst[i]               = (_Float16)Wq[i];
    dst[C_ * C_ + i]     = (_Float16)Wk[i];
    dst[2 * C_ * C_ + i] = (_Float16)Wv[i];
    dst[3 * C_ * C_ + i] = (_Float16)Wo[i];
  }
}

// ---------------------------------------------------------------------------
// Main kernel: one 8x8 window per block.
// ---------------------------------------------------------------------------
__global__ __launch_bounds__(256) void swin_cross_attn_kernel(
    const _Float16* __restrict__ wf16,
    const float* __restrict__ x, const float* __restrict__ y,
    const float* __restrict__ bq, const float* __restrict__ bk,
    const float* __restrict__ bv, const float* __restrict__ bo,
    float* __restrict__ out) {
  extern __shared__ __align__(16) char smem[];
  _Float16* lWq = (_Float16*)(smem + OFF_WQ);
  _Float16* lWk = (_Float16*)(smem + OFF_WK);
  _Float16* lWv = (_Float16*)(smem + OFF_WV);
  _Float16* lWo = (_Float16*)(smem + OFF_WO);
  _Float16* lX  = (_Float16*)(smem + OFF_X);
  _Float16* lY  = (_Float16*)(smem + OFF_Y);
  _Float16* lQ  = (_Float16*)(smem + OFF_Q);
  _Float16* lK  = (_Float16*)(smem + OFF_K);
  _Float16* lVt = (_Float16*)(smem + OFF_VT);
  float*    lS  = (float*)(smem + OFF_S);
  _Float16* lP  = (_Float16*)(smem + OFF_P);

  const int tid  = threadIdx.x;
  // Wave index is wave-uniform: pin it to an SGPR so tile loops stay scalar.
  const int wave = __builtin_amdgcn_readfirstlane(tid >> 5);
  const int lane = tid & 31;
  const int lr   = lane & 15;
  const int blk  = blockIdx.x;
  const int b    = blk >> 10;       // / NWIN_
  const int win  = blk & 1023;
  const int wi   = win >> 5;        // window row
  const int wj   = win & 31;        // window col
  const float SCALE = 0.17677669529663687f;   // hd^-0.5, folded into Q

  // Each lane's output column is wave*16+lr for every 8-col-tile GEMM
  // (t = it*8+wave => nt = wave), so one bias scalar per projection suffices.
  const float biasQ = bq[wave * 16 + lr];
  const float biasK = bk[wave * 16 + lr];
  const float biasV = bv[wave * 16 + lr];
  const float biasO = bo[wave * 16 + lr];

  // ---- Stage A: async-DMA f16 weights (128 KiB) global -> padded LDS ------
  // 8192 16B chunks; issued up-front so the DMA overlaps the window gather.
  for (int it = 0; it < 8; ++it) {
    const int i = it * 256 + tid;
    const int r = i >> 4, c8 = (i & 15) << 3;
    const unsigned loff = (unsigned)((r * SWD + c8) * 2);
    const _Float16* g = wf16 + (size_t)i * 8;
    async_g2l_b128((unsigned)OFF_WQ + loff, g);
    async_g2l_b128((unsigned)OFF_WK + loff, g + 16384);
    async_g2l_b128((unsigned)OFF_WV + loff, g + 32768);
    async_g2l_b128((unsigned)OFF_WO + loff, g + 49152);
  }

  // ---- Stage A2: gather shifted 8x8 windows of x and y -> LDS f16 ---------
  {
    const size_t plane = (size_t)H_ * W_;
    const float* xb = x + (size_t)b * C_ * plane;
    const float* yb = y + (size_t)b * C_ * plane;
#pragma unroll
    for (int it = 0; it < 32; ++it) {
      int i = it * 256 + tid;
      int c = i >> 6, m = i & 63;
      int r = m >> 3, cc = m & 7;
      int hh = (wi * WSZ + r + SHIFT_) & (H_ - 1);   // roll(-4) gather
      int ww = (wj * WSZ + cc + SHIFT_) & (W_ - 1);
      size_t off = (size_t)c * plane + (size_t)hh * W_ + ww;
      lX[(size_t)m * SWD + c] = (_Float16)xb[off];
      lY[(size_t)m * SWD + c] = (_Float16)yb[off];
    }
  }
  wait_asynccnt0();   // weight DMA landed (per-wave), then rendezvous
  __syncthreads();

  // ---- Stage B: projections (64x128x128 each): tile (mt=it, nt=wave) ------
  // Q = X Wq^T + bq (pre-scaled)
#pragma unroll
  for (int it = 0; it < 4; ++it) {
    v8f acc = {0.f, 0.f, 0.f, 0.f, 0.f, 0.f, 0.f, 0.f};
#pragma unroll
    for (int ks = 0; ks < 4; ++ks)
      acc = wmma_f16(ldfrag_rm(lX, it * 16, ks * 32, SWD),
                     ldfrag_rm(lWq, wave * 16, ks * 32, SWD), acc);
    stfrag_f16_bias(lQ, it * 16, wave * 16, SWD, acc, SCALE, biasQ);
  }
  // K = Y Wk^T + bk
#pragma unroll
  for (int it = 0; it < 4; ++it) {
    v8f acc = {0.f, 0.f, 0.f, 0.f, 0.f, 0.f, 0.f, 0.f};
#pragma unroll
    for (int ks = 0; ks < 4; ++ks)
      acc = wmma_f16(ldfrag_rm(lY, it * 16, ks * 32, SWD),
                     ldfrag_rm(lWk, wave * 16, ks * 32, SWD), acc);
    stfrag_f16_bias(lK, it * 16, wave * 16, SWD, acc, 1.0f, biasK);
  }
  // V = Y Wv^T + bv, stored transposed [channel][token] for the P@V B-loads
#pragma unroll
  for (int it = 0; it < 4; ++it) {
    v8f acc = {0.f, 0.f, 0.f, 0.f, 0.f, 0.f, 0.f, 0.f};
#pragma unroll
    for (int ks = 0; ks < 4; ++ks)
      acc = wmma_f16(ldfrag_rm(lY, it * 16, ks * 32, SWD),
                     ldfrag_rm(lWv, wave * 16, ks * 32, SWD), acc);
    stfrag_f16_T(lVt, it * 16, wave * 16, SVT, acc, biasV);
  }
  __syncthreads();

  // ---- Stage C: scores S[h] = Qh Kh^T (64x64 per head, K-dim = 32) --------
#pragma unroll
  for (int it = 0; it < 8; ++it) {
    const int t = it * 8 + wave;
    const int head = t >> 4, rem = t & 15;
    const int mt = rem >> 2, nt = rem & 3;   // nt == wave & 3
    v8f acc = {0.f, 0.f, 0.f, 0.f, 0.f, 0.f, 0.f, 0.f};
    acc = wmma_f16(ldfrag_rm(lQ, mt * 16, head * HD_, SWD),
                   ldfrag_rm(lK, nt * 16, head * HD_, SWD), acc);
    stfrag_f32_nb(lS, head * N_ + mt * 16, nt * 16, SSC, acc);
  }
  __syncthreads();

  // ---- Stage D: softmax over 64 keys; one thread per (head,row) -----------
  {
    float* srow = lS + (size_t)tid * SSC;
    _Float16* prow = lP + (size_t)tid * SAT;
    float mx = -3.4e38f;
#pragma unroll 8
    for (int j = 0; j < 64; ++j) mx = fmaxf(mx, srow[j]);
    float sum = 0.f;
#pragma unroll 8
    for (int j = 0; j < 64; ++j) { float e = __expf(srow[j] - mx); srow[j] = e; sum += e; }
    float inv = 1.0f / sum;
#pragma unroll 8
    for (int j = 0; j < 64; ++j) prow[j] = (_Float16)(srow[j] * inv);
  }
  __syncthreads();

  // ---- Stage E: O[h] = P[h] V[h] (64x32/head, K-dim = 64); O reuses lX ----
  // t = it*8+wave => head = it, mt = wave>>1, nt2 = wave&1 (fixed per wave).
  _Float16* lO = lX;
  {
    const int mt = wave >> 1, nt2 = wave & 1;
#pragma unroll
    for (int head = 0; head < 4; ++head) {
      v8f acc = {0.f, 0.f, 0.f, 0.f, 0.f, 0.f, 0.f, 0.f};
#pragma unroll
      for (int ks = 0; ks < 2; ++ks)
        acc = wmma_f16(ldfrag_rm(lP, head * N_ + mt * 16, ks * 32, SAT),
                       ldfrag_rm(lVt, head * HD_ + nt2 * 16, ks * 32, SVT), acc);
      stfrag_f16_nb(lO, mt * 16, head * HD_ + nt2 * 16, SWD, acc);
    }
  }
  __syncthreads();

  // ---- Stage F: Out = O Wo^T + bo -> staged f32 (reuses score buffer) -----
  float* lOut = lS;   // 64 x SWD f32 fits inside the score region
#pragma unroll
  for (int it = 0; it < 4; ++it) {
    v8f acc = {0.f, 0.f, 0.f, 0.f, 0.f, 0.f, 0.f, 0.f};
#pragma unroll
    for (int ks = 0; ks < 4; ++ks)
      acc = wmma_f16(ldfrag_rm(lO, it * 16, ks * 32, SWD),
                     ldfrag_rm(lWo, wave * 16, ks * 32, SWD), acc);
    stfrag_f32_bias(lOut, it * 16, wave * 16, SWD, acc, biasO);
  }
  __syncthreads();

  // ---- Stage G: async LDS->global scatter with inverse roll ---------------
  {
    const size_t plane = (size_t)H_ * W_;
    const float* ob = out + (size_t)b * C_ * plane;
#pragma unroll
    for (int it = 0; it < 32; ++it) {
      int i = it * 256 + tid;
      int c = i >> 6, m = i & 63;
      int r = m >> 3, cc = m & 7;
      int hh = (wi * WSZ + r + SHIFT_) & (H_ - 1);
      int ww = (wj * WSZ + cc + SHIFT_) & (W_ - 1);
      const float* gp = ob + (size_t)c * plane + (size_t)hh * W_ + ww;
      unsigned loff = (unsigned)(OFF_S + ((size_t)m * SWD + c) * sizeof(float));
      async_l2g_b32(gp, loff);
    }
  }
  wait_asynccnt0();   // drain ASYNCcnt (s_endpgm also waits-idle)
}

// ---------------------------------------------------------------------------
extern "C" void kernel_launch(void* const* d_in, const int* in_sizes, int n_in,
                              void* d_out, int out_size, void* d_ws, size_t ws_size,
                              hipStream_t stream) {
  const float* x  = (const float*)d_in[0];
  const float* y  = (const float*)d_in[1];
  const float* Wq = (const float*)d_in[2];
  const float* bq = (const float*)d_in[3];
  const float* Wk = (const float*)d_in[4];
  const float* bk = (const float*)d_in[5];
  const float* Wv = (const float*)d_in[6];
  const float* bv = (const float*)d_in[7];
  const float* Wo = (const float*)d_in[8];
  const float* bo = (const float*)d_in[9];
  float* out = (float*)d_out;
  _Float16* wf16 = (_Float16*)d_ws;   // 4 * 128*128 f16 = 128 KiB of scratch

  (void)in_sizes; (void)n_in; (void)out_size; (void)ws_size;

  hipFuncSetAttribute((const void*)swin_cross_attn_kernel,
                      hipFuncAttributeMaxDynamicSharedMemorySize,
                      (int)SMEM_TOTAL);

  cvt_weights_kernel<<<dim3((C_ * C_ + 255) / 256), dim3(256), 0, stream>>>(
      Wq, Wk, Wv, Wo, wf16);

  swin_cross_attn_kernel<<<dim3(B_ * NWIN_), dim3(256), SMEM_TOTAL, stream>>>(
      wf16, x, y, bq, bk, bv, bo, out);
}
